// GNNClassifier_69793218560497
// MI455X (gfx1250) — matrix operand order repaired and checked by reading, not compiled, not run on hardware
//
#include <hip/hip_runtime.h>
#include <hip/hip_bf16.h>
#include <math.h>

#define N_NODES 100000   // exact multiple of 16 -> GEMM tiles need no row guards
#define N_EDGES 1600000
#define N_GRAPHS 512
#define HID 128

typedef __attribute__((ext_vector_type(16))) _Float16 v16h;
typedef __attribute__((ext_vector_type(8)))  float    v8f;

// ---------------- degree / dinv / layer-1 scalar aggregation ----------------

__global__ void deg_init_k(float* deg, int n) {
    int i = blockIdx.x * blockDim.x + threadIdx.x;
    if (i < n) deg[i] = 1.0f;  // self-loop
}

__global__ void deg_edge_k(const int* __restrict__ dst, float* deg, int e) {
    int i = blockIdx.x * blockDim.x + threadIdx.x;
    if (i < e) atomicAdd(&deg[dst[i]], 1.0f);
}

__global__ void dinv_sinit_k(float* dinv, const float* __restrict__ x, float* s, int n) {
    int i = blockIdx.x * blockDim.x + threadIdx.x;
    if (i >= n) return;
    float d = rsqrtf(dinv[i]);   // deg >= 1 always
    dinv[i] = d;
    s[i] = d * d * x[i];         // self-loop term of layer-1 scalar aggregate
}

__global__ void s_edge_k(const int* __restrict__ src, const int* __restrict__ dst,
                         const float* __restrict__ x, const float* __restrict__ dinv,
                         float* s, float* norm, int e) {
    int i = blockIdx.x * blockDim.x + threadIdx.x;
    if (i >= e) return;
    int a = src[i], b = dst[i];
    float nv = dinv[a] * dinv[b];
    norm[i] = nv;                       // reused by layer 2
    atomicAdd(&s[b], nv * x[a]);
}

// ---------------- pack w2 (f32 row-major [k][c]) into per-lane B fragments ----------------
// Fragment layout (16-bit B, 32x16): lane<16 -> col=lane, K=0..15 ; lane>=16 -> same col, K=16..31
// pack index: (((t*128 + col)*2 + half)*16 + j), value = w2[(t*32 + half*16 + j)*128 + col]

__global__ void packw2_k(const float* __restrict__ w2, _Float16* w2p) {
    int i = blockIdx.x * blockDim.x + threadIdx.x;
    if (i >= 4 * 128 * 2 * 16) return;
    int j    = i & 15;
    int half = (i >> 4) & 1;
    int col  = (i >> 5) & 127;
    int t    = (i >> 12) & 3;
    int k = t * 32 + half * 16 + j;
    w2p[i] = (_Float16)w2[k * 128 + col];
}

// ---------------- GEMM: hh = relu(s*w1 + b1) @ w2, f16 WMMA, f32 acc ----------------
// block = 256 threads = 8 waves; block handles 16 rows x 128 cols; wave w -> cols [16w,16w+16)
// N_NODES % 16 == 0 -> every tile is full, no guards anywhere.

__global__ void __launch_bounds__(256) gemm_k(const float* __restrict__ s,
                                              const float* __restrict__ w1,
                                              const float* __restrict__ b1,
                                              const _Float16* __restrict__ w2p,
                                              float* __restrict__ hh) {
    __shared__ _Float16 At[16][136];  // padded to dodge bank conflicts
    int rowbase = blockIdx.x * 16;
    int tid = threadIdx.x;

    // build h1 tile in LDS: h1[m][k] = relu(s[row]*w1[k] + b1[k])
    for (int i = tid; i < 16 * 128; i += 256) {
        int m = i >> 7, k = i & 127;
        float sv = s[rowbase + m];
        float hv = fmaxf(sv * w1[k] + b1[k], 0.0f);
        At[m][k] = (_Float16)hv;
    }
    __syncthreads();

    int lane = tid & 31, wv = tid >> 5;
    int m = lane & 15, half = lane >> 4;
    int colbase = wv * 16;

    v8f acc = {};
#pragma unroll
    for (int t = 0; t < 4; ++t) {
        // A fragment: lane<16 holds K = t*32+0..15; lane>=16 holds K = t*32+16..31
        v16h a;
        int kb = t * 32 + half * 16;
#pragma unroll
        for (int j = 0; j < 8; ++j) {
            a[j]     = At[m][kb + j];
            a[8 + j] = At[m][kb + 8 + j];
        }
        // B fragment: contiguous 16 halves from the packed buffer
        const v16h* bp = (const v16h*)(w2p + ((((t * 128 + (colbase + m)) * 2 + half) << 4)));
        v16h b = *bp;
        acc = __builtin_amdgcn_wmma_f32_16x16x32_f16(false, a, false, b,
                                                     (short)0, acc, false, false);
    }

    // D layout: VGPR v -> M = v + 8*half, N = lane&15 ; unguarded straight-line stores
    float* outp = hh + (rowbase + 8 * half) * 128 + colbase + m;
#pragma unroll
    for (int v = 0; v < 8; ++v) {
        outp[v * 128] = acc[v];
    }
}

// ---------------- layer-2 propagation ----------------

__global__ void self2_k(const float* __restrict__ dinv, const float* __restrict__ hh,
                        float* __restrict__ out2, int n) {
    int idx = blockIdx.x * blockDim.x + threadIdx.x;   // n*32 threads
    if (idx >= n * 32) return;
    int node = idx >> 5;
    int c = (idx & 31) * 4;
    float d = dinv[node];
    float dd = d * d;
    float4 v = *(const float4*)(hh + node * 128 + c);
    float4 o = make_float4(dd * v.x, dd * v.y, dd * v.z, dd * v.w);
    *(float4*)(out2 + node * 128 + c) = o;
}

__global__ void edge2_k(const int* __restrict__ src, const int* __restrict__ dst,
                        const float* __restrict__ norm, const float* __restrict__ hh,
                        float* __restrict__ out2, int e) {
    long long idx = (long long)blockIdx.x * blockDim.x + threadIdx.x;  // e*32 threads
    if (idx >= (long long)e * 32) return;
    int ei = (int)(idx >> 5);
    int c = ((int)idx & 31) * 4;
    int a = src[ei], b = dst[ei];
    float nv = norm[ei];
    float4 v = *(const float4*)(hh + a * 128 + c);
    float* o = out2 + b * 128 + c;
    atomicAdd(o + 0, nv * v.x);
    atomicAdd(o + 1, nv * v.y);
    atomicAdd(o + 2, nv * v.z);
    atomicAdd(o + 3, nv * v.w);
}

// ---------------- pooling ----------------

__global__ void zero_pool_k(float* pool, float* cnt) {
    int i = blockIdx.x * blockDim.x + threadIdx.x;
    if (i < N_GRAPHS * HID) pool[i] = 0.0f;
    if (i < N_GRAPHS) cnt[i] = 0.0f;
}

__global__ void pool_k(const float* __restrict__ out2, const float* __restrict__ b2,
                       const int* __restrict__ batch, float* pool, float* cnt, int n) {
    int idx = blockIdx.x * blockDim.x + threadIdx.x;   // n*32 threads
    if (idx >= n * 32) return;
    int node = idx >> 5;
    int c = (idx & 31) * 4;
    int g = batch[node];
    float4 v = *(const float4*)(out2 + node * 128 + c);
    float4 bb = *(const float4*)(b2 + c);
    float* p = pool + g * 128 + c;
    atomicAdd(p + 0, fmaxf(v.x + bb.x, 0.0f));
    atomicAdd(p + 1, fmaxf(v.y + bb.y, 0.0f));
    atomicAdd(p + 2, fmaxf(v.z + bb.z, 0.0f));
    atomicAdd(p + 3, fmaxf(v.w + bb.w, 0.0f));
    if ((idx & 31) == 0) atomicAdd(&cnt[g], 1.0f);
}

// ---------------- MLP head: one block per graph ----------------

__global__ void __launch_bounds__(128) head_k(const float* __restrict__ pool,
                                              const float* __restrict__ cnt,
                                              const float* __restrict__ cw1,
                                              const float* __restrict__ cb1,
                                              const float* __restrict__ cw2,
                                              const float* __restrict__ cb2,
                                              float* __restrict__ out) {
    __shared__ float p[128];
    __shared__ float zpart[4][32];
    __shared__ float z[32];
    int g = blockIdx.x, t = threadIdx.x;
    float inv = 1.0f / fmaxf(cnt[g], 1.0f);
    p[t] = pool[g * 128 + t] * inv;
    __syncthreads();
    int j = t & 31, q = t >> 5;
    float acc = 0.0f;
    for (int c = q * 32; c < q * 32 + 32; ++c) acc += p[c] * cw1[c * 32 + j];
    zpart[q][j] = acc;
    __syncthreads();
    if (t < 32) {
        float v = zpart[0][t] + zpart[1][t] + zpart[2][t] + zpart[3][t] + cb1[t];
        z[t] = fmaxf(v, 0.0f);
    }
    __syncthreads();
    if (t == 0) {
        float acc2 = 0.0f;
        for (int jj = 0; jj < 32; ++jj) acc2 += z[jj] * cw2[jj];
        float v = acc2 + cb2[0];
        out[g] = 1.0f / (1.0f + expf(-v));
    }
}

// ---------------- launcher ----------------

extern "C" void kernel_launch(void* const* d_in, const int* in_sizes, int n_in,
                              void* d_out, int out_size, void* d_ws, size_t ws_size,
                              hipStream_t stream) {
    const float* x    = (const float*)d_in[0];
    const int*   ei   = (const int*)d_in[1];
    const int*   src  = ei;
    const int*   dst  = ei + N_EDGES;
    const int*   batch= (const int*)d_in[2];
    const float* w1   = (const float*)d_in[3];
    const float* b1   = (const float*)d_in[4];
    const float* w2   = (const float*)d_in[5];
    const float* b2   = (const float*)d_in[6];
    const float* cw1  = (const float*)d_in[7];
    const float* cb1  = (const float*)d_in[8];
    const float* cw2  = (const float*)d_in[9];
    const float* cb2  = (const float*)d_in[10];
    float* out = (float*)d_out;

    // workspace layout (floats); offsets kept 16B/32B aligned
    float* ws   = (float*)d_ws;
    float* dinv = ws;                          // N (deg during accumulation)
    float* s    = dinv + 100096;               // N
    float* norm = s + 100096;                  // E
    float* hh   = norm + N_EDGES;              // N*128
    float* out2 = hh + (size_t)N_NODES * HID;  // N*128
    float* pool = out2 + (size_t)N_NODES * HID;// 512*128
    float* cnt  = pool + N_GRAPHS * HID;       // 512
    _Float16* w2p = (_Float16*)(cnt + 512);    // 16384 halves (32KB)

    const int B = 256;

    // degrees (with self-loops) -> dinv, layer-1 scalar aggregate s
    deg_init_k<<<(N_NODES + B - 1) / B, B, 0, stream>>>(dinv, N_NODES);
    deg_edge_k<<<(N_EDGES + B - 1) / B, B, 0, stream>>>(dst, dinv, N_EDGES);
    dinv_sinit_k<<<(N_NODES + B - 1) / B, B, 0, stream>>>(dinv, x, s, N_NODES);
    s_edge_k<<<(N_EDGES + B - 1) / B, B, 0, stream>>>(src, dst, x, dinv, s, norm, N_EDGES);

    // pack w2 into WMMA B fragments, then hh = relu(s*w1+b1) @ w2
    packw2_k<<<(16384 + B - 1) / B, B, 0, stream>>>(w2, w2p);
    gemm_k<<<N_NODES / 16, 256, 0, stream>>>(s, w1, b1, w2p, hh);

    // layer-2 propagation: self-loop init + edge scatter
    self2_k<<<(N_NODES * 32 + B - 1) / B, B, 0, stream>>>(dinv, hh, out2, N_NODES);
    {
        long long total = (long long)N_EDGES * 32;
        int blocks = (int)((total + B - 1) / B);
        edge2_k<<<blocks, B, 0, stream>>>(src, dst, norm, hh, out2, N_EDGES);
    }

    // global mean pool (bias+relu fused), then MLP head with sigmoid
    zero_pool_k<<<(N_GRAPHS * HID + B - 1) / B, B, 0, stream>>>(pool, cnt);
    pool_k<<<(N_NODES * 32 + B - 1) / B, B, 0, stream>>>(out2, b2, batch, pool, cnt, N_NODES);
    head_k<<<N_GRAPHS, 128, 0, stream>>>(pool, cnt, cw1, cb1, cw2, cb2, out);
}
